// HGANConv_59854664237680
// MI455X (gfx1250) — matrix operand order repaired and checked by reading, not compiled, not run on hardware
//
#include <hip/hip_runtime.h>
#include <hip/hip_bf16.h>
#include <math.h>

#define HD 128          // hidden channels H
#define DIN 256         // input feature dim D

// ---------------------------------------------------------------------------
// WMMA types (probe-confirmed signature for gfx1250):
//   __builtin_amdgcn_wmma_f32_16x16x32_bf16(bool, v16bf, bool, v16bf,
//                                           short, v8f, bool, bool)
// ---------------------------------------------------------------------------
typedef __attribute__((ext_vector_type(16))) __bf16 v16bf;
typedef __attribute__((ext_vector_type(8)))  float  v8f;

union AFrag  { unsigned u[8]; v16bf v; };
union BFragP { uint4 q[2]; unsigned u[8]; v16bf v; };

// pack two f32 -> packed bf16 pair (RNE), low = a, high = b
__device__ __forceinline__ unsigned pk_bf16(float a, float b) {
  unsigned ua = __float_as_uint(a);
  unsigned ub = __float_as_uint(b);
  ua += 0x7FFFu + ((ua >> 16) & 1u);
  ub += 0x7FFFu + ((ub >> 16) & 1u);
  return (ua >> 16) | (ub & 0xFFFF0000u);
}

// ordered-uint encoding for float atomic max
__device__ __forceinline__ unsigned f2o(float f) {
  unsigned u = __float_as_uint(f);
  return (u & 0x80000000u) ? ~u : (u | 0x80000000u);
}
__device__ __forceinline__ float o2f(unsigned u) {
  u = (u & 0x80000000u) ? (u & 0x7FFFFFFFu) : ~u;
  return __uint_as_float(u);
}

// ---------------------------------------------------------------------------
// Pre-pack weight matrix B into per-lane WMMA B-fragment layout (bf16 dwords).
// Bpk dword index = ((ntile*NKF + f)*32 + lane)*8 + v
//   lane 0-15 : column n = ntile*16 + lane,  K = f*32 + 16*(v>>2) + 2*(v&3), +1
//   lane 16-31: same column set,             K = f*32 + 8 + 16*(v>>2) + 2*(v&3), +1
//   bT==0 : B stored [K, Nout] row-major   (x @ w case)
//   bT==1 : B stored [Nout, K] row-major   (h @ W^T GRU case)
// ---------------------------------------------------------------------------
__global__ void pack_b(const float* __restrict__ B, unsigned* __restrict__ Bpk,
                       int K, int Nout, int bT) {
  const int NKF   = K >> 5;
  const int total = (Nout >> 4) * NKF * 256;   // dwords
  int idx = blockIdx.x * blockDim.x + threadIdx.x;
  if (idx >= total) return;
  const int v     = idx & 7;
  const int lane  = (idx >> 3) & 31;
  const int fr    = idx >> 8;                  // ntile*NKF + f
  const int ntile = fr / NKF;
  const int f     = fr - ntile * NKF;
  const int ncol  = ntile * 16 + (lane & 15);
  const int k     = f * 32 + (lane >> 4) * 8 + (v & 3) * 2 + (v >> 2) * 16;
  float a, b;
  if (bT) { a = B[(size_t)ncol * K + k];    b = B[(size_t)ncol * K + k + 1]; }
  else    { a = B[(size_t)k * Nout + ncol]; b = B[(size_t)(k + 1) * Nout + ncol]; }
  Bpk[idx] = pk_bf16(a, b);
}

// ---------------------------------------------------------------------------
// GEMM: C[M,Nout] = A[M,K] @ B + bias   (bf16 WMMA, f32 accumulate)
// One wave per 16-row strip; all K/32 A-fragments resident in registers;
// B read as pre-packed fragments (2x b128 per fragment, L2-resident).
// ---------------------------------------------------------------------------
template <int NKF, bool HASB>
__global__ void gemm_wmma_bf16(const float* __restrict__ A,
                               const uint4* __restrict__ Bpk,
                               const float* __restrict__ bias,
                               float* __restrict__ C,
                               int M, int Nout) {
  const int K     = NKF * 32;
  const int lane  = threadIdx.x & 31;
  const int wave  = threadIdx.x >> 5;
  const int strip = blockIdx.x * (blockDim.x >> 5) + wave;
  const int m0    = strip * 16;
  if (m0 >= M) return;

  const int mrow  = lane & 15;   // A row within tile / C column within tile
  const int khalf = lane >> 4;   // lane half selects K sub-range

  // --- load + pack all A fragments for this strip (16x32 bf16 layout) ---
  AFrag afrag[NKF];
#pragma unroll
  for (int f = 0; f < NKF; ++f) {
    const float4* ap = (const float4*)(A + (size_t)(m0 + mrow) * K + f * 32 + khalf * 8);
    float4 a0 = ap[0], a1 = ap[1];      // K = base .. base+7
    float4 a2 = ap[4], a3 = ap[5];      // K = base+16 .. base+23
    afrag[f].u[0] = pk_bf16(a0.x, a0.y);
    afrag[f].u[1] = pk_bf16(a0.z, a0.w);
    afrag[f].u[2] = pk_bf16(a1.x, a1.y);
    afrag[f].u[3] = pk_bf16(a1.z, a1.w);
    afrag[f].u[4] = pk_bf16(a2.x, a2.y);
    afrag[f].u[5] = pk_bf16(a2.z, a2.w);
    afrag[f].u[6] = pk_bf16(a3.x, a3.y);
    afrag[f].u[7] = pk_bf16(a3.z, a3.w);
  }

  // --- sweep output column tiles ---
  const int ntiles = Nout >> 4;
  for (int nt = 0; nt < ntiles; ++nt) {
    const int ncol = nt * 16 + mrow;
    v8f acc = {};
#pragma unroll
    for (int f = 0; f < NKF; ++f) {
      BFragP bf;
      const uint4* bp = Bpk + ((size_t)(nt * NKF + f) * 32 + lane) * 2;
      bf.q[0] = bp[0];
      bf.q[1] = bp[1];
      acc = __builtin_amdgcn_wmma_f32_16x16x32_bf16(
          false, afrag[f].v, false, bf.v, (short)0, acc, false, false);
    }
    // C/D layout: VGPR r -> row m0 + khalf*8 + r, col = ncol
    float badd = 0.0f;
    if (HASB) badd = bias[ncol];
    const int rbase = m0 + khalf * 8;
#pragma unroll
    for (int r = 0; r < 8; ++r)
      C[(size_t)(rbase + r) * Nout + ncol] = acc[r] + badd;
  }
}

// ---------------------------------------------------------------------------
// alpha_src[n] = hs[n,:] . q[0:H] ;  alpha_tgt[n] = ht[n,:] . q[H:2H]
// one wave per node, shuffle reduction
// ---------------------------------------------------------------------------
__global__ void alpha_kernel(const float* __restrict__ hs,
                             const float* __restrict__ ht,
                             const float* __restrict__ q,
                             float* __restrict__ asrc,
                             float* __restrict__ atgt, int N) {
  const int node = (int)((blockIdx.x * (long long)blockDim.x + threadIdx.x) >> 5);
  const int lane = threadIdx.x & 31;
  if (node >= N) return;
  float s = 0.f, t = 0.f;
#pragma unroll
  for (int j = lane; j < HD; j += 32) {
    s += hs[(size_t)node * HD + j] * q[j];
    t += ht[(size_t)node * HD + j] * q[HD + j];
  }
#pragma unroll
  for (int off = 16; off > 0; off >>= 1) {
    s += __shfl_xor(s, off, 32);
    t += __shfl_xor(t, off, 32);
  }
  if (lane == 0) { asrc[node] = s; atgt[node] = t; }
}

__global__ void init_seg(unsigned* __restrict__ segmax, float* __restrict__ den, int N) {
  int i = blockIdx.x * blockDim.x + threadIdx.x;
  if (i < N) { segmax[i] = 0x007FFFFFu /* f2o(-inf) */; den[i] = 0.f; }
}

__global__ void fill_zero_f(float* __restrict__ p, long long n) {
  long long i = blockIdx.x * (long long)blockDim.x + threadIdx.x;
  if (i < n) p[i] = 0.f;
}

// logit[e] = leaky_relu(asrc[si]+atgt[ti], 0.2); segment max via ordered-uint
__global__ void edge_logits_max(const int* __restrict__ si, const int* __restrict__ ti,
                                const float* __restrict__ asrc,
                                const float* __restrict__ atgt,
                                float* __restrict__ logit,
                                unsigned* __restrict__ segmax, int E) {
  int e = blockIdx.x * blockDim.x + threadIdx.x;
  if (e >= E) return;
  float x = asrc[si[e]] + atgt[ti[e]];
  x = x > 0.f ? x : 0.2f * x;
  logit[e] = x;
  atomicMax(&segmax[ti[e]], f2o(x));
}

// decode ordered-uint max, neutralize empty (-inf) segments -> 0
__global__ void seg_finalize(unsigned* __restrict__ segmax, int N) {
  int i = blockIdx.x * blockDim.x + threadIdx.x;
  if (i >= N) return;
  float v = o2f(segmax[i]);
  if (!isfinite(v)) v = 0.f;
  ((float*)segmax)[i] = v;   // in-place reuse as float m[]
}

__global__ void edge_exp_sum(const int* __restrict__ ti,
                             const float* __restrict__ logit,
                             const float* __restrict__ m,
                             float* __restrict__ ex, float* __restrict__ den, int E) {
  int e = blockIdx.x * blockDim.x + threadIdx.x;
  if (e >= E) return;
  int t = ti[e];
  float v = __expf(logit[e] - m[t]);
  ex[e] = v;
  atomicAdd(&den[t], v);
}

// agg[ti,:] += hs[si,:] * ex[e]/(den[ti]+1e-16)   (128 threads per edge;
// e is lane-uniform -> ti/ex/den loads become scalar)
__global__ void edge_aggregate(const int* __restrict__ si, const int* __restrict__ ti,
                               const float* __restrict__ hs,
                               const float* __restrict__ ex,
                               const float* __restrict__ den,
                               float* __restrict__ agg, long long total) {
  long long idx = blockIdx.x * (long long)blockDim.x + threadIdx.x;
  if (idx >= total) return;
  int e = (int)(idx >> 7);
  int j = (int)(idx & 127);
  int t = ti[e];
  float att = ex[e] / (den[t] + 1e-16f);
  atomicAdd(&agg[(size_t)t * HD + j], hs[(size_t)si[e] * HD + j] * att);
}

// GRUCell elementwise: input=ht path (gi), hidden=agg path (gh)
__global__ void gru_kernel(const float* __restrict__ gi, const float* __restrict__ gh,
                           const float* __restrict__ agg, float* __restrict__ out,
                           long long total, int do_relu) {
  long long idx = blockIdx.x * (long long)blockDim.x + threadIdx.x;
  if (idx >= total) return;
  int n = (int)(idx >> 7);
  int j = (int)(idx & 127);
  const float* gin = gi + (size_t)n * 3 * HD;
  const float* ghn = gh + (size_t)n * 3 * HD;
  float r  = 1.f / (1.f + __expf(-(gin[j] + ghn[j])));
  float z  = 1.f / (1.f + __expf(-(gin[HD + j] + ghn[HD + j])));
  float nn = tanhf(gin[2 * HD + j] + r * ghn[2 * HD + j]);
  float v  = (1.f - z) * nn + z * agg[idx];
  out[idx] = do_relu ? fmaxf(v, 0.f) : v;
}

// semantic attention over two item relations + relu; wave per node
__global__ void sem_att_item(const float* __restrict__ h0, const float* __restrict__ h1,
                             const float* __restrict__ g, float* __restrict__ out, int N) {
  const int node = (int)((blockIdx.x * (long long)blockDim.x + threadIdx.x) >> 5);
  const int lane = threadIdx.x & 31;
  if (node >= N) return;
  float l0 = 0.f, l1 = 0.f;
#pragma unroll
  for (int j = lane; j < HD; j += 32) {
    float gv = g[j];
    l0 += h0[(size_t)node * HD + j] * gv;
    l1 += h1[(size_t)node * HD + j] * gv;
  }
#pragma unroll
  for (int off = 16; off > 0; off >>= 1) {
    l0 += __shfl_xor(l0, off, 32);
    l1 += __shfl_xor(l1, off, 32);
  }
  float e0 = __expf(l0), e1 = __expf(l1);
  float inv = 1.f / (e0 + e1);
  float a0 = e0 * inv, a1 = e1 * inv;
#pragma unroll
  for (int j = lane; j < HD; j += 32) {
    float v = a0 * h0[(size_t)node * HD + j] + a1 * h1[(size_t)node * HD + j];
    out[(size_t)node * HD + j] = fmaxf(v, 0.f);
  }
}

// ---------------------------------------------------------------------------
// host driver
// ---------------------------------------------------------------------------
static inline int cdiv(long long a, long long b) { return (int)((a + b - 1) / b); }

extern "C" void kernel_launch(void* const* d_in, const int* in_sizes, int n_in,
                              void* d_out, int out_size, void* d_ws, size_t ws_size,
                              hipStream_t stream) {
  (void)n_in; (void)out_size; (void)ws_size;
  const int N = in_sizes[0] / DIN;

  const float* x_user = (const float*)d_in[0];
  const float* x_item = (const float*)d_in[1];
  const int*   eiArr[3] = {(const int*)d_in[2], (const int*)d_in[3], (const int*)d_in[4]};
  int Ecnt[3] = {in_sizes[2] / 2, in_sizes[3] / 2, in_sizes[4] / 2};
  int Emax = Ecnt[0]; if (Ecnt[1] > Emax) Emax = Ecnt[1]; if (Ecnt[2] > Emax) Emax = Ecnt[2];
  const float* g_item = (const float*)d_in[27];   // g_user (26) unused: R=1 softmax == identity

  float* out_user = (float*)d_out;                       // [N*H]
  float* out_item = (float*)d_out + (size_t)N * HD;      // [N*H]

  // workspace carve (16B-aligned granularity throughout)
  float* p        = (float*)d_ws;
  unsigned* Bpk   = (unsigned*)p; p += 32768;            // packed-B scratch (max 24576 dwords)
  float* hs       = p; p += (size_t)N * HD;
  float* ht       = p; p += (size_t)N * HD;
  float* agg      = p; p += (size_t)N * HD;
  float* gi       = p; p += (size_t)N * 3 * HD;
  float* gh       = p; p += (size_t)N * 3 * HD;
  float* asrc     = p; p += N;
  float* atgt     = p; p += N;
  unsigned* segm  = (unsigned*)p; p += N;                // ordered-max, then m[] in-place
  float* den      = p; p += N;
  float* logit    = p; p += Emax;
  float* ex       = p; p += Emax;
  float* h_ui     = p; p += (size_t)N * HD;              // h_item_ui
  float* h_ii     = p; p += (size_t)N * HD;              // h_item_ii

  const int strips      = N / 16;                        // N % 16 == 0 for N=50000
  const int gemm_blocks = cdiv(strips, 8);               // 8 waves / block
  const long long NH    = (long long)N * HD;
  const int pkXW  = (HD / 16) * (DIN / 32) * 256;        // pack sizes (dwords)
  const int pkGRU = ((3 * HD) / 16) * (HD / 32) * 256;

  for (int r = 0; r < 3; ++r) {
    const float* x_src  = (r == 0) ? x_user : x_item;
    const float* x_tgt  = (r == 1) ? x_user : x_item;
    const int*   si     = eiArr[r];
    const int*   ti     = eiArr[r] + Ecnt[r];
    const int    E      = Ecnt[r];
    const int    base   = 5 + 7 * r;
    const float* w_src  = (const float*)d_in[base + 0];
    const float* w_tgt  = (const float*)d_in[base + 1];
    const float* q      = (const float*)d_in[base + 2];
    const float* wih    = (const float*)d_in[base + 3];
    const float* whh    = (const float*)d_in[base + 4];
    const float* bih    = (const float*)d_in[base + 5];
    const float* bhh    = (const float*)d_in[base + 6];
    float*       h_dst  = (r == 0) ? h_ui : (r == 1) ? out_user : h_ii;
    const int    relu   = (r == 1);                      // user sem-att is identity+relu

    // node linears: hs = x_src @ w_src, ht = x_tgt @ w_tgt  (K=256 -> NKF=8)
    pack_b<<<cdiv(pkXW, 256), 256, 0, stream>>>(w_src, Bpk, DIN, HD, 0);
    gemm_wmma_bf16<8, false><<<gemm_blocks, 256, 0, stream>>>(
        x_src, (const uint4*)Bpk, nullptr, hs, N, HD);
    pack_b<<<cdiv(pkXW, 256), 256, 0, stream>>>(w_tgt, Bpk, DIN, HD, 0);
    gemm_wmma_bf16<8, false><<<gemm_blocks, 256, 0, stream>>>(
        x_tgt, (const uint4*)Bpk, nullptr, ht, N, HD);

    // edge attention (q-split decomposition)
    alpha_kernel<<<cdiv((long long)N * 32, 256), 256, 0, stream>>>(hs, ht, q, asrc, atgt, N);
    init_seg<<<cdiv(N, 256), 256, 0, stream>>>(segm, den, N);
    fill_zero_f<<<cdiv(NH, 256), 256, 0, stream>>>(agg, NH);
    edge_logits_max<<<cdiv(E, 256), 256, 0, stream>>>(si, ti, asrc, atgt, logit, segm, E);
    seg_finalize<<<cdiv(N, 256), 256, 0, stream>>>(segm, N);
    edge_exp_sum<<<cdiv(E, 256), 256, 0, stream>>>(ti, logit, (const float*)segm, ex, den, E);
    edge_aggregate<<<cdiv((long long)E * HD, 256), 256, 0, stream>>>(si, ti, hs, ex, den, agg,
                                                                     (long long)E * HD);

    // GRUCell: gi = ht @ wih^T + bih ; gh = agg @ whh^T + bhh   (K=128 -> NKF=4)
    pack_b<<<cdiv(pkGRU, 256), 256, 0, stream>>>(wih, Bpk, HD, 3 * HD, 1);
    gemm_wmma_bf16<4, true><<<gemm_blocks, 256, 0, stream>>>(
        ht, (const uint4*)Bpk, bih, gi, N, 3 * HD);
    pack_b<<<cdiv(pkGRU, 256), 256, 0, stream>>>(whh, Bpk, HD, 3 * HD, 1);
    gemm_wmma_bf16<4, true><<<gemm_blocks, 256, 0, stream>>>(
        agg, (const uint4*)Bpk, bhh, gh, N, 3 * HD);
    gru_kernel<<<cdiv(NH, 256), 256, 0, stream>>>(gi, gh, agg, h_dst, NH, relu);
  }

  // item semantic attention over {h_item_ui, h_item_ii} + relu
  sem_att_item<<<cdiv((long long)N * 32, 256), 256, 0, stream>>>(h_ui, h_ii, g_item, out_item, N);
}
